// PointNetPlusPlus_89988154786549
// MI455X (gfx1250) — compile-verified, matrix operand order checked by reading, not run on hardware
//
#include <hip/hip_runtime.h>

typedef __attribute__((ext_vector_type(16))) _Float16 v16h;
typedef __attribute__((ext_vector_type(8)))  float    v8f;

union V16H { v16h v; unsigned u[8]; };

// K-pair base for WMMA f16 A/B fragments (ISA 7.12.2):
// lane<16: K pairs {0,2,4,6,16,18,20,22}; lane>=16: {8,10,12,14,24,26,28,30}
__device__ __forceinline__ int kbase_for(int lane, int vv) {
  int h = (lane >> 4) & 1;
  return (vv < 4) ? (8 * h + 2 * vv) : (16 + 8 * h + 2 * (vv - 4));
}

#define WMMA_F16(A, B, C) \
  __builtin_amdgcn_wmma_f32_16x16x32_f16(false, (A), false, (B), (short)0, (C), false, false)

// ---------------------------------------------------------------------------
// Weight packing: f32 [fi,fo] -> f16 padded [Kp,Np] with consecutive-K pairs
// interleaved so ((k>>1)*Np + n) as u32 gives the (k,k+1) pair for column n.
// ---------------------------------------------------------------------------
__global__ __launch_bounds__(256) void pack_w_kernel(
    const float* __restrict__ w, int fi, int fo, int Kp, int Np,
    _Float16* __restrict__ dst)
{
  int e = blockIdx.x * blockDim.x + threadIdx.x;
  if (e >= Kp * Np) return;
  int k = e / Np, n = e - k * Np;
  float v = (k < fi && n < fo) ? w[(size_t)k * fo + n] : 0.0f;
  dst[((size_t)(k >> 1) * Np + n) * 2 + (k & 1)] = (_Float16)v;
}

__global__ __launch_bounds__(256) void zero_f16_kernel(_Float16* dst, int count) {
  int i = blockIdx.x * blockDim.x + threadIdx.x;
  if (i < count) dst[i] = (_Float16)0.0f;
}

__global__ __launch_bounds__(256) void copy_cols_f16_kernel(
    const float* __restrict__ src, int C,
    _Float16* __restrict__ dst, int Kp, int coloff, int M)
{
  int e = blockIdx.x * blockDim.x + threadIdx.x;
  if (e >= M * C) return;
  int m = e / C, c = e - m * C;
  dst[(size_t)m * Kp + coloff + c] = (_Float16)src[(size_t)m * C + c];
}

// ---------------------------------------------------------------------------
// Farthest point sampling: one 1024-thread block per cloud, serial scan with
// register-resident dmin and shuffle+LDS argmax reduction. Gathers pos too.
// ---------------------------------------------------------------------------
__global__ __launch_bounds__(1024) void fps_kernel(
    const float* __restrict__ pos_src, int Nin, int nOut,
    int* __restrict__ idx_out, float* __restrict__ pos_out)
{
  int bb = blockIdx.x;
  const float* P = pos_src + (size_t)bb * Nin * 3;
  int* io = idx_out + (size_t)bb * nOut;
  float* po = pos_out + (size_t)bb * nOut * 3;

  __shared__ float lx, ly, lz;
  __shared__ float wval[32];
  __shared__ int   widx[32];

  int tid = threadIdx.x;
  float dmin[8];
#pragma unroll
  for (int j = 0; j < 8; ++j) dmin[j] = 1e30f;

  if (tid == 0) {
    io[0] = 0;
    lx = P[0]; ly = P[1]; lz = P[2];
    po[0] = lx; po[1] = ly; po[2] = lz;
  }
  __syncthreads();

  for (int i = 1; i < nOut; ++i) {
    float bx = lx, by = ly, bz = lz;
    float bestv = -1.0f; int bestp = 0x7fffffff;
#pragma unroll
    for (int j = 0; j < 8; ++j) {
      int p = tid + j * 1024;
      if (p < Nin) {
        float dx = P[p * 3 + 0] - bx, dy = P[p * 3 + 1] - by, dz = P[p * 3 + 2] - bz;
        float d = dx * dx + dy * dy + dz * dz;
        float dm = fminf(dmin[j], d);
        dmin[j] = dm;
        if (dm > bestv || (dm == bestv && p < bestp)) { bestv = dm; bestp = p; }
      }
    }
    for (int off = 16; off > 0; off >>= 1) {
      float ov = __shfl_xor(bestv, off, 32);
      int   oi = __shfl_xor(bestp, off, 32);
      if (ov > bestv || (ov == bestv && oi < bestp)) { bestv = ov; bestp = oi; }
    }
    int w = tid >> 5;
    if ((tid & 31) == 0) { wval[w] = bestv; widx[w] = bestp; }
    __syncthreads();
    if (tid == 0) {
      float bv = wval[0]; int bi = widx[0];
      for (int w2 = 1; w2 < 32; ++w2)
        if (wval[w2] > bv || (wval[w2] == bv && widx[w2] < bi)) { bv = wval[w2]; bi = widx[w2]; }
      io[i] = bi;
      lx = P[bi * 3 + 0]; ly = P[bi * 3 + 1]; lz = P[bi * 3 + 2];
      po[i * 3 + 0] = lx; po[i * 3 + 1] = ly; po[i * 3 + 2] = lz;
    }
    __syncthreads();
  }
}

// ---------------------------------------------------------------------------
// Ball-query top-64: distances in LDS, 64 iterated parallel argmins.
// Set semantics suffice (masked max aggregation is order-invariant).
// ---------------------------------------------------------------------------
__global__ __launch_bounds__(256) void ball_topk_kernel(
    const float* __restrict__ pos_q, const float* __restrict__ pos_src,
    int n, int Nin, float r2,
    int* __restrict__ nb, unsigned char* __restrict__ valid)
{
  extern __shared__ float smem_bq[];
  float* d2s = smem_bq;              // [Nin]
  float* redv = d2s + Nin;           // [8]
  int*   redi = (int*)(redv + 8);    // [8]

  int q = blockIdx.x;
  int bb = q / n, qi = q - bb * n;
  const float* P = pos_src + (size_t)bb * Nin * 3;
  const float* Q = pos_q + ((size_t)bb * n + qi) * 3;
  int* nbq = nb + ((size_t)bb * n + qi) * 64;
  unsigned char* vq = valid + ((size_t)bb * n + qi) * 64;
  float qx = Q[0], qy = Q[1], qz = Q[2];
  int tid = threadIdx.x;

  for (int p = tid; p < Nin; p += blockDim.x) {
    float dx = P[p * 3] - qx, dy = P[p * 3 + 1] - qy, dz = P[p * 3 + 2] - qz;
    d2s[p] = dx * dx + dy * dy + dz * dz;
  }
  __syncthreads();

  for (int s = 0; s < 64; ++s) {
    float bv = 3.0e38f; int bi = 0;
    for (int p = tid; p < Nin; p += blockDim.x) {
      float dv = d2s[p];
      if (dv < bv || (dv == bv && p < bi)) { bv = dv; bi = p; }
    }
    for (int off = 16; off > 0; off >>= 1) {
      float ov = __shfl_xor(bv, off, 32);
      int   oi = __shfl_xor(bi, off, 32);
      if (ov < bv || (ov == bv && oi < bi)) { bv = ov; bi = oi; }
    }
    int w = tid >> 5;
    if ((tid & 31) == 0) { redv[w] = bv; redi[w] = bi; }
    __syncthreads();
    if (tid == 0) {
      for (int w2 = 1; w2 < 8; ++w2)
        if (redv[w2] < bv || (redv[w2] == bv && redi[w2] < bi)) { bv = redv[w2]; bi = redi[w2]; }
      nbq[s] = bi;
      vq[s] = (bv <= r2) ? 1 : 0;
      d2s[bi] = 3.0e38f;
    }
    __syncthreads();
  }
}

// ---------------------------------------------------------------------------
// Fused SA PointNetConv: one workgroup (8 waves) per query point.
// Build X(64 x fi_pad) in LDS -> WMMA GEMM1 (+bias,ReLU) -> H1 in LDS ->
// WMMA GEMM2 (+bias) -> masked max over 64 neighbor rows in registers.
// Each wave owns a 64x16 output strip: one B fragment feeds 4 WMMAs.
// ---------------------------------------------------------------------------
__global__ __launch_bounds__(256) void sa_fused_kernel(
    const float* __restrict__ x_src,   // [B, Nin, Cin]
    const float* __restrict__ pos_src, // [B, Nin, 3]
    const float* __restrict__ pos_q,   // [B, n, 3]
    const int*   __restrict__ nb,      // [B, n, 64]
    const unsigned char* __restrict__ valid, // [B, n, 64]
    const unsigned* __restrict__ W1,   // packed [fi_pad/2, d1] u32 pairs
    const float* __restrict__ b1,
    const unsigned* __restrict__ W2,   // packed [d1/2, d2]
    const float* __restrict__ b2,
    float* __restrict__ out,           // [B, n, d2]
    int n, int Nin, int Cin, int fi_pad, int d1, int d2)
{
  extern __shared__ char smem_sa[];
  _Float16* Xs  = (_Float16*)smem_sa;        // [64, fi_pad]
  _Float16* H1s = Xs + 64 * fi_pad;          // [64, d1]
  float* pq = (float*)(H1s + 64 * d1);       // [4]
  unsigned char* vld = (unsigned char*)(pq + 4); // [64]

  int q = blockIdx.x;
  int bb = q / n, qi = q - bb * n;
  const float* xs = x_src + (size_t)bb * Nin * Cin;
  const float* ps = pos_src + (size_t)bb * Nin * 3;
  const float* pqg = pos_q + ((size_t)bb * n + qi) * 3;
  const int* nbq = nb + ((size_t)bb * n + qi) * 64;
  const unsigned char* vq = valid + ((size_t)bb * n + qi) * 64;
  float* outq = out + ((size_t)bb * n + qi) * d2;

  int tid = threadIdx.x;
  if (tid < 3) pq[tid] = pqg[tid];
  if (tid < 64) vld[tid] = vq[tid];
  __syncthreads();

  // Build X = concat(x[nb], pos[nb]-pos_q), zero padded to fi_pad
  int fi = Cin + 3;
  for (int e = tid; e < 64 * fi_pad; e += blockDim.x) {
    int row = e / fi_pad, c = e - row * fi_pad;
    float val = 0.0f;
    int g = nbq[row];
    if (c < Cin)     val = xs[(size_t)g * Cin + c];
    else if (c < fi) val = ps[(size_t)g * 3 + (c - Cin)] - pq[c - Cin];
    Xs[row * fi_pad + c] = (_Float16)val;
  }
  __syncthreads();

  int wave = tid >> 5, lane = tid & 31;
  int m0 = lane & 15;
  const unsigned* Xs32 = (const unsigned*)Xs;
  int ldX2 = fi_pad >> 1;

  // GEMM1: H1 = relu(X @ W1 + b1), 64 x d1 (one wave per 16-col strip)
  int ct1 = d1 >> 4;
  for (int c = wave; c < ct1; c += 8) {
    int col = (c << 4) + (lane & 15);
    v8f acc0 = {}, acc1 = {}, acc2 = {}, acc3 = {};
    for (int k0 = 0; k0 < fi_pad; k0 += 32) {
      V16H bm, a0, a1, a2, a3;
#pragma unroll
      for (int vv = 0; vv < 8; ++vv) {
        int o = (k0 + kbase_for(lane, vv)) >> 1;
        bm.u[vv] = W1[(size_t)o * d1 + col];
        a0.u[vv] = Xs32[(m0     ) * ldX2 + o];
        a1.u[vv] = Xs32[(m0 + 16) * ldX2 + o];
        a2.u[vv] = Xs32[(m0 + 32) * ldX2 + o];
        a3.u[vv] = Xs32[(m0 + 48) * ldX2 + o];
      }
      acc0 = WMMA_F16(a0.v, bm.v, acc0);
      acc1 = WMMA_F16(a1.v, bm.v, acc1);
      acc2 = WMMA_F16(a2.v, bm.v, acc2);
      acc3 = WMMA_F16(a3.v, bm.v, acc3);
    }
    float bias = b1[col];
#pragma unroll
    for (int i = 0; i < 8; ++i) {
      int r = i + ((lane >> 4) << 3);
      float v0 = acc0[i] + bias; if (v0 < 0.0f) v0 = 0.0f;
      float v1 = acc1[i] + bias; if (v1 < 0.0f) v1 = 0.0f;
      float v2 = acc2[i] + bias; if (v2 < 0.0f) v2 = 0.0f;
      float v3 = acc3[i] + bias; if (v3 < 0.0f) v3 = 0.0f;
      H1s[(r     ) * d1 + col] = (_Float16)v0;
      H1s[(r + 16) * d1 + col] = (_Float16)v1;
      H1s[(r + 32) * d1 + col] = (_Float16)v2;
      H1s[(r + 48) * d1 + col] = (_Float16)v3;
    }
  }
  __syncthreads();

  // GEMM2 + masked max aggregation over the 64 neighbor rows
  const unsigned* H132 = (const unsigned*)H1s;
  int ldH2 = d1 >> 1;
  int ct2 = d2 >> 4;
  for (int c = wave; c < ct2; c += 8) {
    int col = (c << 4) + (lane & 15);
    v8f acc0 = {}, acc1 = {}, acc2 = {}, acc3 = {};
    for (int k0 = 0; k0 < d1; k0 += 32) {
      V16H bm, a0, a1, a2, a3;
#pragma unroll
      for (int vv = 0; vv < 8; ++vv) {
        int o = (k0 + kbase_for(lane, vv)) >> 1;
        bm.u[vv] = W2[(size_t)o * d2 + col];
        a0.u[vv] = H132[(m0     ) * ldH2 + o];
        a1.u[vv] = H132[(m0 + 16) * ldH2 + o];
        a2.u[vv] = H132[(m0 + 32) * ldH2 + o];
        a3.u[vv] = H132[(m0 + 48) * ldH2 + o];
      }
      acc0 = WMMA_F16(a0.v, bm.v, acc0);
      acc1 = WMMA_F16(a1.v, bm.v, acc1);
      acc2 = WMMA_F16(a2.v, bm.v, acc2);
      acc3 = WMMA_F16(a3.v, bm.v, acc3);
    }
    float bias = b2[col];
    float colmax = -3.0e38f;
#pragma unroll
    for (int i = 0; i < 8; ++i) {
      int r = i + ((lane >> 4) << 3);
      colmax = fmaxf(colmax, vld[r     ] ? (acc0[i] + bias) : -1e30f);
      colmax = fmaxf(colmax, vld[r + 16] ? (acc1[i] + bias) : -1e30f);
      colmax = fmaxf(colmax, vld[r + 32] ? (acc2[i] + bias) : -1e30f);
      colmax = fmaxf(colmax, vld[r + 48] ? (acc3[i] + bias) : -1e30f);
    }
    float other = __shfl_xor(colmax, 16, 32);
    colmax = fmaxf(colmax, other);
    if (lane < 16) outq[col] = (colmax <= -1e30f) ? 0.0f : colmax;
  }
}

// ---------------------------------------------------------------------------
// 3-NN inverse-distance interpolation, writes f16 into the FP input matrix.
// ---------------------------------------------------------------------------
__global__ __launch_bounds__(256) void knn3_kernel(
    const float* __restrict__ x_src,   // [B,S,C]
    const float* __restrict__ pos_src, // [B,S,3]
    const float* __restrict__ pos_tgt, // [B,T,3]
    int S, int T, int C, int Kp, int total,
    _Float16* __restrict__ dst)        // [B,T,Kp], cols [0,C)
{
  int t = blockIdx.x * blockDim.x + threadIdx.x;
  if (t >= total) return;
  int bb = t / T;
  const float* P = pos_src + (size_t)bb * S * 3;
  const float* X = x_src + (size_t)bb * S * C;
  const float* Q = pos_tgt + (size_t)t * 3;
  float qx = Q[0], qy = Q[1], qz = Q[2];
  float bd0 = 3e38f, bd1 = 3e38f, bd2 = 3e38f;
  int bi0 = 0, bi1 = 0, bi2 = 0;
  for (int s = 0; s < S; ++s) {
    float dx = P[s * 3] - qx, dy = P[s * 3 + 1] - qy, dz = P[s * 3 + 2] - qz;
    float d = dx * dx + dy * dy + dz * dz;
    if (d < bd0)      { bd2 = bd1; bi2 = bi1; bd1 = bd0; bi1 = bi0; bd0 = d; bi0 = s; }
    else if (d < bd1) { bd2 = bd1; bi2 = bi1; bd1 = d; bi1 = s; }
    else if (d < bd2) { bd2 = d; bi2 = s; }
  }
  float w0 = 1.0f / fmaxf(bd0, 1e-16f);
  float w1 = 1.0f / fmaxf(bd1, 1e-16f);
  float w2 = 1.0f / fmaxf(bd2, 1e-16f);
  float inv = 1.0f / (w0 + w1 + w2);
  _Float16* drow = dst + (size_t)t * Kp;
  const float* r0 = X + (size_t)bi0 * C;
  const float* r1 = X + (size_t)bi1 * C;
  const float* r2 = X + (size_t)bi2 * C;
  for (int c = 0; c < C; ++c)
    drow[c] = (_Float16)((w0 * r0[c] + w1 * r1[c] + w2 * r2[c]) * inv);
}

// ---------------------------------------------------------------------------
// Generic WMMA GEMM: one wave per 64x16 output strip (M always multiple of
// 64 here), so each B (weight) fragment feeds 4 WMMAs.
// A f16 row-major [M,Kp]; Wp packed f16 [Kp/2,Np] u32 pairs; out f32 or f16.
// ---------------------------------------------------------------------------
__global__ __launch_bounds__(256) void gemm_kernel(
    const _Float16* __restrict__ A,
    const unsigned* __restrict__ Wp,
    const float* __restrict__ bias,
    int M, int Kp, int Np, int Nc,
    float* __restrict__ outF, _Float16* __restrict__ outH,
    int ldO, int relu)
{
  int wave = threadIdx.x >> 5, lane = threadIdx.x & 31;
  int tilesN = (Nc + 15) >> 4;
  long tiles = (long)(M >> 6) * tilesN;
  long tile = (long)blockIdx.x * 8 + wave;
  if (tile >= tiles) return;
  int tm = (int)(tile / tilesN), tn = (int)(tile - (long)tm * tilesN);
  int m0 = tm * 64 + (lane & 15);
  int col = tn * 16 + (lane & 15);
  const unsigned* A32 = (const unsigned*)A;
  int ldA2 = Kp >> 1;
  v8f acc0 = {}, acc1 = {}, acc2 = {}, acc3 = {};
  for (int k0 = 0; k0 < Kp; k0 += 32) {
    V16H bm, a0, a1, a2, a3;
#pragma unroll
    for (int vv = 0; vv < 8; ++vv) {
      int o = (k0 + kbase_for(lane, vv)) >> 1;
      bm.u[vv] = Wp[(size_t)o * Np + col];
      a0.u[vv] = A32[(size_t)(m0     ) * ldA2 + o];
      a1.u[vv] = A32[(size_t)(m0 + 16) * ldA2 + o];
      a2.u[vv] = A32[(size_t)(m0 + 32) * ldA2 + o];
      a3.u[vv] = A32[(size_t)(m0 + 48) * ldA2 + o];
    }
    acc0 = WMMA_F16(a0.v, bm.v, acc0);
    acc1 = WMMA_F16(a1.v, bm.v, acc1);
    acc2 = WMMA_F16(a2.v, bm.v, acc2);
    acc3 = WMMA_F16(a3.v, bm.v, acc3);
  }
  float bv = (col < Nc) ? bias[col] : 0.0f;
#pragma unroll
  for (int i = 0; i < 8; ++i) {
    int r = tm * 64 + i + ((lane >> 4) << 3);
    float v0 = acc0[i] + bv;
    float v1 = acc1[i] + bv;
    float v2 = acc2[i] + bv;
    float v3 = acc3[i] + bv;
    if (relu) {
      v0 = fmaxf(v0, 0.0f); v1 = fmaxf(v1, 0.0f);
      v2 = fmaxf(v2, 0.0f); v3 = fmaxf(v3, 0.0f);
    }
    if (col < Nc) {
      if (outF) {
        outF[(size_t)(r     ) * ldO + col] = v0;
        outF[(size_t)(r + 16) * ldO + col] = v1;
        outF[(size_t)(r + 32) * ldO + col] = v2;
        outF[(size_t)(r + 48) * ldO + col] = v3;
      }
      if (outH) {
        outH[(size_t)(r     ) * ldO + col] = (_Float16)v0;
        outH[(size_t)(r + 16) * ldO + col] = (_Float16)v1;
        outH[(size_t)(r + 32) * ldO + col] = (_Float16)v2;
        outH[(size_t)(r + 48) * ldO + col] = (_Float16)v3;
      }
    }
  }
}

// ---------------------------------------------------------------------------
// Host orchestration
// ---------------------------------------------------------------------------
static void launch_gemm(const _Float16* A, const _Float16* Wp, const float* bias,
                        int M, int Kp, int Np, int Nc,
                        float* outF, _Float16* outH, int ldO, int relu,
                        hipStream_t s)
{
  int tilesN = (Nc + 15) >> 4;
  long tiles = (long)(M >> 6) * tilesN;
  int blocks = (int)((tiles + 7) / 8);
  gemm_kernel<<<blocks, 256, 0, s>>>(A, (const unsigned*)Wp, bias,
                                     M, Kp, Np, Nc, outF, outH, ldO, relu);
}

extern "C" void kernel_launch(void* const* d_in, const int* in_sizes, int n_in,
                              void* d_out, int out_size, void* d_ws, size_t ws_size,
                              hipStream_t stream)
{
  (void)n_in; (void)out_size; (void)ws_size;
  const int Bc = 4, N = 8192;
  const float* x   = (const float*)d_in[0];
  const float* pos = (const float*)d_in[1];

  // module dims: 0..3 sa1..sa4, 4..7 fp4..fp1, 8 sem, 9 inst
  struct MDm { int d0, d1, d2; };
  const MDm md[10] = {{7,64,64},{67,128,128},{131,256,256},{259,512,512},
                      {768,256,256},{384,128,128},{192,64,64},{68,64,64},
                      {64,64,8},{64,64,64}};
  // Identify param groups by unique (|w1|,|w2|) signature (robust to dict order).
  const float *w1[10], *b1[10], *w2[10], *b2[10];
  for (int g = 0; g < 10; ++g) {
    int base = 3 + 4 * g;
    long s1 = in_sizes[base], s2 = in_sizes[base + 2];
    for (int m = 0; m < 10; ++m) {
      if ((long)md[m].d0 * md[m].d1 == s1 && (long)md[m].d1 * md[m].d2 == s2) {
        w1[m] = (const float*)d_in[base + 0]; b1[m] = (const float*)d_in[base + 1];
        w2[m] = (const float*)d_in[base + 2]; b2[m] = (const float*)d_in[base + 3];
        break;
      }
    }
  }

  char* ws = (char*)d_ws;
  size_t cur = 0;
  auto alloc = [&](size_t bytes) -> void* {
    void* p = ws + cur;
    cur += (bytes + 255) & ~(size_t)255;
    return p;
  };
  auto pad16 = [](int v) { return (v + 15) & ~15; };
  auto pad32 = [](int v) { return (v + 31) & ~31; };

  _Float16 *wp1[10], *wp2[10];
  int Kp1[10], Np1[10], Kp2[10], Np2[10];
  for (int m = 0; m < 10; ++m) {
    Kp1[m] = pad32(md[m].d0); Np1[m] = pad16(md[m].d1);
    Kp2[m] = pad32(md[m].d1); Np2[m] = pad16(md[m].d2);
    wp1[m] = (_Float16*)alloc((size_t)Kp1[m] * Np1[m] * 2);
    wp2[m] = (_Float16*)alloc((size_t)Kp2[m] * Np2[m] * 2);
  }
  for (int m = 0; m < 10; ++m) {
    int t1 = Kp1[m] * Np1[m];
    pack_w_kernel<<<(t1 + 255) / 256, 256, 0, stream>>>(w1[m], md[m].d0, md[m].d1, Kp1[m], Np1[m], wp1[m]);
    int t2 = Kp2[m] * Np2[m];
    pack_w_kernel<<<(t2 + 255) / 256, 256, 0, stream>>>(w2[m], md[m].d1, md[m].d2, Kp2[m], Np2[m], wp2[m]);
  }

  const int nL[4] = {2048, 512, 128, 32};
  const float r2L[4] = {0.01f, 0.04f, 0.16f, 0.64f};
  const int fiPad[4] = {32, 96, 160, 288};

  int* idxL[4]; float* posL[4]; int* nbL[4]; unsigned char* vaL[4]; float* xL[4];
  for (int l = 0; l < 4; ++l) {
    idxL[l] = (int*)alloc((size_t)Bc * nL[l] * 4);
    posL[l] = (float*)alloc((size_t)Bc * nL[l] * 3 * 4);
    nbL[l]  = (int*)alloc((size_t)Bc * nL[l] * 64 * 4);
    vaL[l]  = (unsigned char*)alloc((size_t)Bc * nL[l] * 64);
    xL[l]   = (float*)alloc((size_t)Bc * nL[l] * md[l].d2 * 4);
  }

  // ---- SA levels ----
  for (int l = 0; l < 4; ++l) {
    int Nin = (l == 0) ? N : nL[l - 1];
    const float* psrc = (l == 0) ? pos : posL[l - 1];
    const float* xsrc = (l == 0) ? x : xL[l - 1];
    int Cin = (l == 0) ? 4 : md[l - 1].d2;
    int n = nL[l], d1 = md[l].d1, d2 = md[l].d2, fpd = fiPad[l];

    fps_kernel<<<Bc, 1024, 0, stream>>>(psrc, Nin, n, idxL[l], posL[l]);

    size_t shBall = (size_t)Nin * 4 + 32 + 32 + 64;
    ball_topk_kernel<<<Bc * n, 256, shBall, stream>>>(posL[l], psrc, n, Nin, r2L[l], nbL[l], vaL[l]);

    size_t shSA = ((size_t)64 * fpd + (size_t)64 * d1) * 2 + 16 + 64 + 64;
    sa_fused_kernel<<<Bc * n, 256, shSA, stream>>>(
        xsrc, psrc, posL[l], nbL[l], vaL[l],
        (const unsigned*)wp1[l], b1[l], (const unsigned*)wp2[l], b2[l],
        xL[l], n, Nin, Cin, fpd, d1, d2);
  }

  // ---- FP levels: j=0..3 -> modules 4..7 (fp4..fp1) ----
  const int Tj[4]  = {128, 512, 2048, 8192};
  const int Kpj[4] = {768, 384, 192, 96};
  _Float16* fpIn[4]; _Float16* fpH1[4]; float* fpOut[4];
  for (int j = 0; j < 4; ++j) {
    int m = 4 + j;
    fpIn[j]  = (_Float16*)alloc((size_t)Bc * Tj[j] * Kpj[j] * 2);
    fpH1[j]  = (_Float16*)alloc((size_t)Bc * Tj[j] * md[m].d1 * 2);
    fpOut[j] = (float*)alloc((size_t)Bc * Tj[j] * md[m].d2 * 4);
  }
  for (int j = 0; j < 4; ++j) {
    int m = 4 + j;
    const float* src = (j == 0) ? xL[3] : fpOut[j - 1];
    int C = (j == 0) ? md[3].d2 : md[4 + j - 1].d2;
    int S = (j == 0) ? nL[3] : Tj[j - 1];
    const float* ps = posL[3 - j];
    const float* pt = (j == 3) ? pos : posL[2 - j];
    int T = Tj[j];
    const float* skip = (j == 3) ? x : xL[2 - j];
    int Cs = (j == 3) ? 4 : md[2 - j].d2;
    int Kp = Kpj[j];
    int total = Bc * T;

    if (Kp > C + Cs) {
      int cnt = total * Kp;
      zero_f16_kernel<<<(cnt + 255) / 256, 256, 0, stream>>>(fpIn[j], cnt);
    }
    knn3_kernel<<<(total + 255) / 256, 256, 0, stream>>>(src, ps, pt, S, T, C, Kp, total, fpIn[j]);
    copy_cols_f16_kernel<<<(total * Cs + 255) / 256, 256, 0, stream>>>(skip, Cs, fpIn[j], Kp, C, total);

    launch_gemm(fpIn[j], wp1[m], b1[m], total, Kp, Np1[m], md[m].d1,
                nullptr, fpH1[j], md[m].d1, 1, stream);
    launch_gemm(fpH1[j], wp2[m], b2[m], total, Kp2[m], Np2[m], md[m].d2,
                fpOut[j], nullptr, md[m].d2, 0, stream);
  }

  // ---- Heads ----
  int total = Bc * N;
  _Float16* headIn = (_Float16*)alloc((size_t)total * 64 * 2);
  _Float16* semH1  = (_Float16*)alloc((size_t)total * 64 * 2);
  _Float16* instH1 = (_Float16*)alloc((size_t)total * 64 * 2);
  copy_cols_f16_kernel<<<(total * 64 + 255) / 256, 256, 0, stream>>>(fpOut[3], 64, headIn, 64, 0, total);

  float* outSem  = (float*)d_out;
  float* outInst = (float*)d_out + (size_t)total * 8;

  launch_gemm(headIn, wp1[8], b1[8], total, 64, 64, 64, nullptr, semH1, 64, 1, stream);
  launch_gemm(semH1,  wp2[8], b2[8], total, 64, 16, 8,  outSem, nullptr, 8,  0, stream);
  launch_gemm(headIn, wp1[9], b1[9], total, 64, 64, 64, nullptr, instH1, 64, 1, stream);
  launch_gemm(instH1, wp2[9], b2[9], total, 64, 64, 64, outInst, nullptr, 64, 0, stream);
}
